// OCRHead_90701119357016
// MI455X (gfx1250) — matrix operand order repaired and checked by reading, not compiled, not run on hardware
//
#include <hip/hip_runtime.h>
#include <hip/hip_bf16.h>

// ---------------------------------------------------------------------------
// OCR head for MI455X (gfx1250, wave32). HBM-bound (~470MB traffic at
// 23.3 TB/s => ~20us floor). GEMMs on v_wmma_f32_16x16x32_bf16; conv1 (w1 1x1)
// algebraically folded: conv1(ocr@attn2) = (w1@ocr)@attn2. featbf+res2bf
// (134MB) are L2-resident. conv2's shared A matrix (w2) is staged per-tap into
// LDS by the Tensor Data Mover (TENSOR_LOAD_TO_LDS), using TDM's pad feature
// (interval=1024B, amount=8B -> 2064B row pitch) for conflict-free LDS banks.
// ---------------------------------------------------------------------------

#define Bsz 4
#define Cch 512
#define Hh 128
#define Ww 128
#define HWp (Hh*Ww)          // 16384
#define Kc 19
#define KP 32                // K padded to 32 for WMMA N-tiles
#define KDIM (2*Cch*9)       // 9216 contraction length of the 3x3 conv
#define TILE_PITCH 2064      // 2048B row + 2x8B TDM pad -> LDS bank stride 4

typedef __attribute__((ext_vector_type(16))) __bf16 v16bf;
typedef __attribute__((ext_vector_type(8)))  __bf16 v8bf;
typedef __attribute__((ext_vector_type(2)))  __bf16 v2bf;
typedef __attribute__((ext_vector_type(8)))  float  v8f;
typedef __attribute__((ext_vector_type(4)))  unsigned int v4u;
typedef __attribute__((ext_vector_type(8)))  int v8i;
typedef __attribute__((ext_vector_type(4)))  int v4i;

static __device__ inline v8f wmma_bf16(v16bf a, v16bf b, v8f c) {
  // D = A(16x32 bf16) * B(32x16 bf16) + C(16x16 f32)
  return __builtin_amdgcn_wmma_f32_16x16x32_bf16(
      false, a, false, b, (short)0, c, false, false);
}

// Combine the two contiguous 8-element runs of an A-fragment (ISA 7.12.2:
// lane-half h reads K = {8h..8h+7} and {16+8h..16+8h+7}; this file's runs: K = {16h..16h+7} and {16h+8..16h+15}).
static __device__ inline v16bf a_frag(v8bf lo, v8bf hi) {
  return __builtin_shufflevector(lo, hi, 0,1,2,3,4,5,6,7,8,9,10,11,12,13,14,15);
}

// ---------------------------------------------------------------------------
// K0: zero atomic accumulators; build tap-major padded bf16 W2; feature->bf16
// w2bfT[kk][tap*1024 + c2] = w2[kk][c2][tap] (kk>=19 rows are zero).
// ---------------------------------------------------------------------------
__global__ void prep_kernel(float* __restrict__ ocrP,
                            const float* __restrict__ w2,
                            __bf16* __restrict__ w2bfT,
                            const float* __restrict__ feat,
                            __bf16* __restrict__ featbf) {
  int tid = blockIdx.x * blockDim.x + threadIdx.x;
  int stride = gridDim.x * blockDim.x;
  for (int i = tid; i < Bsz * Cch * KP; i += stride) ocrP[i] = 0.f;
  for (int i = tid; i < KP * KDIM; i += stride) {
    int kk = i / KDIM, kidx = i - kk * KDIM;
    int t = kidx >> 10, c2 = kidx & 1023;
    float v = (kk < Kc) ? w2[(size_t)kk * KDIM + c2 * 9 + t] : 0.f;
    w2bfT[i] = (__bf16)v;
  }
  const float2* fin = (const float2*)feat;
  v2bf* fout = (v2bf*)featbf;
  for (int i = tid; i < (Bsz * Cch * HWp) / 2; i += stride) {
    float2 v = fin[i];
    v2bf o = {(__bf16)v.x, (__bf16)v.y};
    fout[i] = o;
  }
}

// ---------------------------------------------------------------------------
// K1: softmax over HW per (b,k); bf16 output, 32-row padded (rows>=19 zero)
// ---------------------------------------------------------------------------
__global__ void ctx_softmax_kernel(const float* __restrict__ ctx,
                                   __bf16* __restrict__ attnbf) {
  int b = blockIdx.x / KP, k = blockIdx.x % KP;
  __bf16* out = attnbf + ((size_t)b * KP + k) * HWp;
  if (k >= Kc) {
    for (int i = threadIdx.x; i < HWp; i += 256) out[i] = (__bf16)0.f;
    return;
  }
  const float* in = ctx + ((size_t)b * Kc + k) * HWp;
  __shared__ float red[256];
  float m = -INFINITY;
  for (int i = threadIdx.x; i < HWp; i += 256) m = fmaxf(m, in[i]);
  red[threadIdx.x] = m; __syncthreads();
  for (int s = 128; s > 0; s >>= 1) {
    if (threadIdx.x < s) red[threadIdx.x] = fmaxf(red[threadIdx.x], red[threadIdx.x + s]);
    __syncthreads();
  }
  m = red[0]; __syncthreads();
  float sum = 0.f;
  for (int i = threadIdx.x; i < HWp; i += 256) sum += __expf(in[i] - m);
  red[threadIdx.x] = sum; __syncthreads();
  for (int s = 128; s > 0; s >>= 1) {
    if (threadIdx.x < s) red[threadIdx.x] += red[threadIdx.x + s];
    __syncthreads();
  }
  float inv = 1.0f / red[0];
  for (int i = threadIdx.x; i < HWp; i += 256) out[i] = (__bf16)(__expf(in[i] - m) * inv);
}

// ---------------------------------------------------------------------------
// K2: ocrP[b][c][k] = sum_n attn[b][k][n] * feat[b][c][n]
// WMMA M=c tile, N=k (2 tiles), K=n; split-K over HW via grid.y + f32 atomics.
// ---------------------------------------------------------------------------
__global__ void ocr_kernel(const __bf16* __restrict__ featbf,
                           const __bf16* __restrict__ attnbf,
                           float* __restrict__ ocrP) {
  int wave = threadIdx.x >> 5, lane = threadIdx.x & 31;
  int b = blockIdx.x / (Cch / 64);
  int cbase = (blockIdx.x % (Cch / 64)) * 64 + wave * 16;
  int n0 = blockIdx.y * 2048;
  int m = lane & 15, half = lane >> 4, kcol = lane & 15;
  const __bf16* frow  = featbf + ((size_t)b * Cch + cbase + m) * HWp;
  const __bf16* arow0 = attnbf + ((size_t)b * KP + kcol) * HWp;
  const __bf16* arow1 = attnbf + ((size_t)b * KP + kcol + 16) * HWp;
  v8f acc0 = {}, acc1 = {};
  for (int nc = 0; nc < 2048; nc += 32) {
    int nb = n0 + nc;
    v8bf alo = *(const v8bf*)&frow[nb + 16 * half];
    v8bf ahi = *(const v8bf*)&frow[nb + 16 * half + 8];
    v16bf a  = a_frag(alo, ahi);
    v16bf b0 = *(const v16bf*)&arow0[nb + 16 * half];
    v16bf b1 = *(const v16bf*)&arow1[nb + 16 * half];
    acc0 = wmma_bf16(a, b0, acc0);
    acc1 = wmma_bf16(a, b1, acc1);
  }
#pragma unroll
  for (int r = 0; r < 8; r++) {
    int c = cbase + r + 8 * half;
    atomicAdd(&ocrP[((size_t)b * Cch + c) * KP + kcol],      acc0[r]);
    atomicAdd(&ocrP[((size_t)b * Cch + c) * KP + kcol + 16], acc1[r]);
  }
}

// ---------------------------------------------------------------------------
// K3: ocrP (f32 [b][c][KP]) -> ocrbfT (bf16 [b][KP][c])
// ---------------------------------------------------------------------------
__global__ void ocr2bf_kernel(const float* __restrict__ ocrP,
                              __bf16* __restrict__ ocrbfT) {
  int i = blockIdx.x * blockDim.x + threadIdx.x;   // B*KP*Cch threads
  int b = i / (KP * Cch);
  int k = (i / Cch) % KP;
  int c = i % Cch;
  ocrbfT[i] = (__bf16)ocrP[((size_t)b * Cch + c) * KP + k];
}

// ---------------------------------------------------------------------------
// K4: relP[b][n][k] = sum_c feat[b][c][n] * ocr[b][c][k]  (f32 out for stats)
// ---------------------------------------------------------------------------
__global__ void rel_kernel(const __bf16* __restrict__ featbf,
                           const __bf16* __restrict__ ocrbfT,
                           float* __restrict__ relP) {
  int wave = threadIdx.x >> 5, lane = threadIdx.x & 31;
  int gw = blockIdx.x * 8 + wave;
  int b = gw / (HWp / 16);
  int nbase = (gw % (HWp / 16)) * 16;
  int m = lane & 15, half = lane >> 4, kcol = lane & 15;
  const __bf16* fcol  = featbf + (size_t)b * Cch * HWp + nbase + m;
  const __bf16* brow0 = ocrbfT + ((size_t)b * KP + kcol) * Cch;
  const __bf16* brow1 = ocrbfT + ((size_t)b * KP + kcol + 16) * Cch;
  v8f acc0 = {}, acc1 = {};
  for (int c0 = 0; c0 < Cch; c0 += 32) {
    v16bf a;
#pragma unroll
    for (int r = 0; r < 8; r++) {        // A: K=c gathers (stride HWp)
      int base = c0 + 2 * r + 16 * half;
      a[2 * r]     = fcol[(size_t)base * HWp];
      a[2 * r + 1] = fcol[(size_t)(base + 1) * HWp];
    }
    v16bf b0 = *(const v16bf*)&brow0[c0 + 16 * half];
    v16bf b1 = *(const v16bf*)&brow1[c0 + 16 * half];
    acc0 = wmma_bf16(a, b0, acc0);
    acc1 = wmma_bf16(a, b1, acc1);
  }
#pragma unroll
  for (int r = 0; r < 8; r++) {
    int n = nbase + r + 8 * half;
    relP[((size_t)b * HWp + n) * KP + kcol]      = acc0[r];
    relP[((size_t)b * HWp + n) * KP + kcol + 16] = acc1[r];
  }
}

// ---------------------------------------------------------------------------
// K5: per (b,k<19): max over HW and 1/sum(exp) for the second softmax
// ---------------------------------------------------------------------------
__global__ void rel_stats_kernel(const float* __restrict__ relP,
                                 float* __restrict__ stats) {
  int b = blockIdx.x / Kc, k = blockIdx.x % Kc;
  __shared__ float red[256];
  float m = -INFINITY;
  for (int n = threadIdx.x; n < HWp; n += 256)
    m = fmaxf(m, relP[((size_t)b * HWp + n) * KP + k]);
  red[threadIdx.x] = m; __syncthreads();
  for (int s = 128; s > 0; s >>= 1) {
    if (threadIdx.x < s) red[threadIdx.x] = fmaxf(red[threadIdx.x], red[threadIdx.x + s]);
    __syncthreads();
  }
  m = red[0]; __syncthreads();
  float sum = 0.f;
  for (int n = threadIdx.x; n < HWp; n += 256)
    sum += __expf(relP[((size_t)b * HWp + n) * KP + k] - m);
  red[threadIdx.x] = sum; __syncthreads();
  for (int s = 128; s > 0; s >>= 1) {
    if (threadIdx.x < s) red[threadIdx.x] += red[threadIdx.x + s];
    __syncthreads();
  }
  if (threadIdx.x == 0) {
    stats[(b * Kc + k) * 2]     = m;
    stats[(b * Kc + k) * 2 + 1] = 1.0f / red[0];
  }
}

// ---------------------------------------------------------------------------
// K6: fold w1 into ocr: wocr[b][co][k] = sum_ci w1[co][ci] * ocrP[b][ci][k]
// ---------------------------------------------------------------------------
__global__ void wocr_kernel(const float* __restrict__ w1,
                            const float* __restrict__ ocrP,
                            float* __restrict__ wocr) {
  int gw = blockIdx.x * 8 + (threadIdx.x >> 5);
  int lane = threadIdx.x & 31;
  int b = gw / Cch, co = gw % Cch;
  float s = 0.f;
  for (int ci = 0; ci < Cch; ci++)
    s += w1[(size_t)co * Cch + ci] * ocrP[((size_t)b * Cch + ci) * KP + lane];
  wocr[((size_t)b * Cch + co) * KP + lane] = s;
}

// ---------------------------------------------------------------------------
// K7: res2bf[b][c][n] = bf16( b1[c] + sum_k wocr[b][c][k]*softmax(rel)[k][n] )
// ---------------------------------------------------------------------------
__global__ void res2_kernel(const float* __restrict__ relP,
                            const float* __restrict__ stats,
                            const float* __restrict__ wocr,
                            const float* __restrict__ b1,
                            __bf16* __restrict__ res2bf) {
  int gw = blockIdx.x * 8 + (threadIdx.x >> 5);
  int lane = threadIdx.x & 31;
  int b = gw / (HWp / 64);
  int n0 = (gw % (HWp / 64)) * 64 + 2 * lane;
  float a2x[Kc], a2y[Kc];
#pragma unroll
  for (int k = 0; k < Kc; k++) {
    float mx  = stats[(b * Kc + k) * 2];
    float inv = stats[(b * Kc + k) * 2 + 1];
    a2x[k] = __expf(relP[((size_t)b * HWp + n0) * KP + k] - mx) * inv;
    a2y[k] = __expf(relP[((size_t)b * HWp + n0 + 1) * KP + k] - mx) * inv;
  }
  for (int c = 0; c < Cch; c++) {
    const float* wr = wocr + ((size_t)b * Cch + c) * KP;
    float s0 = b1[c], s1 = s0;
#pragma unroll
    for (int k = 0; k < Kc; k++) {
      float w = wr[k];
      s0 = fmaf(w, a2x[k], s0);
      s1 = fmaf(w, a2y[k], s1);
    }
    v2bf o = {(__bf16)s0, (__bf16)s1};
    *(v2bf*)&res2bf[((size_t)b * Cch + c) * HWp + n0] = o;
  }
}

// ---------------------------------------------------------------------------
// K8: 3x3 conv over cat=[res2bf, featbf] as tap-major im2col WMMA GEMM.
// M = out-channel (19 pad 32), N = 16 pixels/wave, K = 9216.
// Per tap, the 32x1024 bf16 A slab (64KB) is DMA'd into LDS once per block by
// the TDM (TENSOR_LOAD_TO_LDS). TDM padding (8B per 1024B) gives a 2064B row
// pitch -> row-to-row LDS bank stride of 4 -> the 16 rows of a b128 fragment
// read hit all 64 banks exactly once (conflict-free ds_load_b128).
// ---------------------------------------------------------------------------
__global__ void conv2_kernel(const __bf16* __restrict__ res2bf,
                             const __bf16* __restrict__ featbf,
                             const __bf16* __restrict__ w2bfT,
                             const float* __restrict__ b2,
                             float* __restrict__ out) {
  __shared__ char tileA[KP * TILE_PITCH];          // 66048 B
  int wave = threadIdx.x >> 5, lane = threadIdx.x & 31;
  int gw = blockIdx.x * 8 + wave;
  int b = gw / (HWp / 16);
  int nbase = (gw % (HWp / 16)) * 16;     // 16 consecutive pixels, one row
  int m = lane & 15, half = lane >> 4;
  int y = nbase / Ww, px = (nbase % Ww) + (lane & 15);
  const __bf16* res2b_ = res2bf + (size_t)b * Cch * HWp;
  const __bf16* featb_ = featbf + (size_t)b * Cch * HWp;
  const __bf16 z = (__bf16)0.f;
  v8f acc0 = {}, acc1 = {};
#pragma unroll 1
  for (int t = 0; t < 9; t++) {
    __syncthreads();                       // previous slab fully consumed
    if (wave == 0) {
      // --- TDM descriptor: 2D tile, 1024 x 32 rows of 2B, row stride 9216 ---
      unsigned long long ga =
          (unsigned long long)(uintptr_t)(w2bfT + (size_t)t * 1024);
      unsigned lds = (unsigned)(uintptr_t)(void*)&tileA[0];
      v4u g0;
      g0[0] = 1u;                                         // count=1 (valid)
      g0[1] = lds;                                        // lds_addr
      g0[2] = (unsigned)(ga & 0xffffffffu);               // global_addr[31:0]
      g0[3] = (unsigned)((ga >> 32) & 0x01ffffffu)        // global_addr[56:32]
              | 0x80000000u;                              // type=2 (image)
      v8i g1;
      g1[0] = (1 << 16)                                   // data_size = 2B
            | (1 << 20)                                   // pad_enable
            | (7 << 22)                                   // pad_interval: 256 DW
            | (1 << 25);                                  // pad_amount: 2 DW
      g1[1] = (int)((KDIM & 0xffff) << 16);               // tensor_dim0[15:0]
      g1[2] = (int)((KDIM >> 16) & 0xffff)                // tensor_dim0[31:16]
            | (KP << 16);                                 // tensor_dim1[15:0]
      g1[3] = 0                                           // tensor_dim1[31:16]
            | (1024 << 16);                               // tile_dim0
      g1[4] = KP;                                         // tile_dim1 (dim2=0)
      g1[5] = KDIM;                                       // dim0_stride[31:0]
      g1[6] = 0;                                          // stride hi, dim1_stride lo
      g1[7] = 0;
      v4i zero4 = {0, 0, 0, 0};
      v8i zero8 = {0, 0, 0, 0, 0, 0, 0, 0};
      __builtin_amdgcn_tensor_load_to_lds(g0, g1, zero4, zero4, zero8, 0);
      __builtin_amdgcn_s_wait_tensorcnt(0);
    }
    __syncthreads();                       // slab visible to all waves

    int yy = y + t / 3 - 1, xx = px + t % 3 - 1;
    bool valid = ((unsigned)yy < (unsigned)Hh) && ((unsigned)xx < (unsigned)Ww);
    int soff = valid ? (yy * Ww + xx) : 0;          // clamped in-bounds
    for (int cb = 0; cb < 2 * Cch; cb += 32) {
      const __bf16* plane = (cb < Cch) ? res2b_ + (size_t)cb * HWp
                                       : featb_ + (size_t)(cb - Cch) * HWp;
      // A fragments from LDS (ds_load_b128, conflict-free via TDM padding)
      int coff = 2 * (cb + 16 * half) + 8 * (cb >> 9);  // byte col offset
      const char* r0 = tileA + (size_t)m * TILE_PITCH + coff;
      const char* r1 = tileA + (size_t)(m + 16) * TILE_PITCH + coff;
      v16bf a0 = a_frag(*(const v8bf*)r0, *(const v8bf*)(r0 + 16));
      v16bf a1 = a_frag(*(const v8bf*)r1, *(const v8bf*)(r1 + 16));
      // B fragment: im2col column, 16 channel rows, hoisted tap mask
      const __bf16* pB = plane + (size_t)(16 * half) * HWp + soff;
      __builtin_prefetch(pB + 32 * (size_t)HWp, 0, 0);   // next channel slab
      v16bf bfr;
#pragma unroll
      for (int r = 0; r < 8; r++) {
        __bf16 e0 = pB[(size_t)(2 * r) * HWp];
        __bf16 e1 = pB[(size_t)(2 * r + 1) * HWp];
        bfr[2 * r]     = valid ? e0 : z;
        bfr[2 * r + 1] = valid ? e1 : z;
      }
      acc0 = wmma_bf16(a0, bfr, acc0);
      acc1 = wmma_bf16(a1, bfr, acc1);
    }
  }
#pragma unroll
  for (int r = 0; r < 8; r++) {
    int kk0 = r + 8 * half;      // tile0 rows < 16 < 19: always valid
    int kk1 = kk0 + 16;
    int pix = nbase + (lane & 15);
    out[((size_t)b * Kc + kk0) * HWp + pix] = acc0[r] + b2[kk0];
    if (kk1 < Kc)
      out[((size_t)b * Kc + kk1) * HWp + pix] = acc1[r] + b2[kk1];
  }
}

// ---------------------------------------------------------------------------
// Launcher
// ---------------------------------------------------------------------------
extern "C" void kernel_launch(void* const* d_in, const int* in_sizes, int n_in,
                              void* d_out, int out_size, void* d_ws, size_t ws_size,
                              hipStream_t stream) {
  const float* context = (const float*)d_in[0];  // [B,K,H,W]
  const float* feature = (const float*)d_in[1];  // [B,C,H,W]
  const float* w1      = (const float*)d_in[2];  // [C,C,1,1]
  const float* b1      = (const float*)d_in[3];  // [C]
  const float* w2      = (const float*)d_in[4];  // [K,2C,3,3]
  const float* b2      = (const float*)d_in[5];  // [K]
  float* out = (float*)d_out;                    // [B,K,H,W]

  // workspace layout (256B aligned regions), ~141 MB total
  char* base = (char*)d_ws;
  size_t off = 0;
  auto take = [&](size_t bytes) {
    char* p = base + off;
    off = (off + bytes + 255) & ~(size_t)255;
    return p;
  };
  __bf16* attnbf = (__bf16*)take((size_t)Bsz * KP * HWp * 2);   // 4 MB
  float*  relP   = (float*)take((size_t)Bsz * HWp * KP * 4);    // 8 MB
  float*  ocrP   = (float*)take((size_t)Bsz * Cch * KP * 4);    // 256 KB
  __bf16* ocrbfT = (__bf16*)take((size_t)Bsz * KP * Cch * 2);   // 128 KB
  float*  wocr   = (float*)take((size_t)Bsz * Cch * KP * 4);    // 256 KB
  float*  stats  = (float*)take((size_t)Bsz * Kc * 2 * 4);      // tiny
  __bf16* w2bfT  = (__bf16*)take((size_t)KP * KDIM * 2);        // 576 KB
  __bf16* featbf = (__bf16*)take((size_t)Bsz * Cch * HWp * 2);  // 64 MB
  __bf16* res2bf = (__bf16*)take((size_t)Bsz * Cch * HWp * 2);  // 64 MB
  (void)ws_size; (void)in_sizes; (void)n_in; (void)out_size;

  prep_kernel<<<1024, 256, 0, stream>>>(ocrP, w2, w2bfT, feature, featbf);
  ctx_softmax_kernel<<<Bsz * KP, 256, 0, stream>>>(context, attnbf);
  ocr_kernel<<<dim3(Bsz * (Cch / 64), HWp / 2048), 128, 0, stream>>>(featbf, attnbf, ocrP);
  ocr2bf_kernel<<<(Bsz * KP * Cch) / 256, 256, 0, stream>>>(ocrP, ocrbfT);
  wocr_kernel<<<(Bsz * Cch) / 8, 256, 0, stream>>>(w1, ocrP, wocr);
  rel_kernel<<<(Bsz * HWp / 16) / 8, 256, 0, stream>>>(featbf, ocrbfT, relP);
  rel_stats_kernel<<<Bsz * Kc, 256, 0, stream>>>(relP, stats);
  res2_kernel<<<(Bsz * HWp / 64) / 8, 256, 0, stream>>>(relP, stats, wocr, b1, res2bf);
  conv2_kernel<<<(Bsz * HWp / 16) / 8, 256, 0, stream>>>(res2bf, featbf, w2bfT, b2, out);
}
